// CustomMaskedLSTM_31275951849674
// MI455X (gfx1250) — matrix operand order repaired and checked by reading, not compiled, not run on hardware
//
#include <hip/hip_runtime.h>

// ---------------- problem constants ----------------
#define T_LEN   2048
#define BATCH   64
#define ISZ     256
#define HID     512
#define GATES   (4 * HID)          // 2048 output columns (i,f,g,o)
#define KTOT    (ISZ + HID)        // 768 fused K
#define KCH     (KTOT / 32)        // 24 WMMA K-chunks
#define NBLOCKS 32                 // one workgroup per channel-tile
#define NTHREADS 128               // 4 waves: one per batch-tile

typedef _Float16 v16h __attribute__((ext_vector_type(16)));
typedef _Float16 v8h  __attribute__((ext_vector_type(8)));
typedef float    v8f  __attribute__((ext_vector_type(8)));

// ---------------- workspace layout (bytes) ----------------
#define WC_BYTES ((size_t)GATES * KTOT * 2)           // fused f16 weights, K-major per column: 3 MB
#define XH_BYTES ((size_t)T_LEN * BATCH * ISZ * 2)    // f16 copy of x: 64 MB
#define HB_BYTES ((size_t)2 * BATCH * HID * 2)        // ping-pong masked h (f16): 128 KB
#define WC_OFF   ((size_t)0)
#define XH_OFF   (WC_OFF + WC_BYTES)
#define HB_OFF   (XH_OFF + XH_BYTES)
#define SY_OFF   (HB_OFF + HB_BYTES)                  // sync counters

#if __has_builtin(__builtin_amdgcn_sched_group_barrier)
#define SCHED_PIPE2()                                           \
    do {                                                        \
        __builtin_amdgcn_sched_group_barrier(0x100, 8, 0);      \
        __builtin_amdgcn_sched_group_barrier(0x020, 2, 0);      \
        __builtin_amdgcn_sched_group_barrier(0x008, 4, 0);      \
        __builtin_amdgcn_sched_group_barrier(0x100, 8, 0);      \
        __builtin_amdgcn_sched_group_barrier(0x020, 2, 0);      \
        __builtin_amdgcn_sched_group_barrier(0x008, 4, 0);      \
    } while (0)
#else
#define SCHED_PIPE2() do {} while (0)
#endif

// ---------------- helpers ----------------
__device__ __forceinline__ float sigf(float x) {
    return 1.0f / (1.0f + __expf(-x));
}
__device__ __forceinline__ float tanh_fast(float x) {
    float t = __expf(-2.0f * fabsf(x));          // in (0,1], no overflow
    float r = (1.0f - t) / (1.0f + t);
    return copysignf(r, x);
}
// concatenate two 8xf16 16-byte segments into a v16h WMMA operand
__device__ __forceinline__ v16h ld2(const _Float16* p0, const _Float16* p1) {
    v8h lo = *(const v8h*)p0;
    v8h hi = *(const v8h*)p1;
    v16h r;
#pragma unroll
    for (int e = 0; e < 8; ++e) { r[e] = lo[e]; r[8 + e] = hi[e]; }
    return r;
}
// A-operand pointer for chunk kn (uniform select between x and h regions)
__device__ __forceinline__ const _Float16* aptr(const _Float16* xrow, const _Float16* hrow, int kn) {
    return (kn < 8) ? (xrow + kn * 32) : (hrow + (kn - 8) * 32);
}

__device__ __forceinline__ void grid_sync(volatile unsigned* cnt, volatile unsigned* gen) {
    __threadfence();                 // make this block's h writes device-visible
    __syncthreads();
    if (threadIdx.x == 0) {
        unsigned g = *gen;
        if (atomicAdd((unsigned*)cnt, 1u) == NBLOCKS - 1) {
            *cnt = 0;
            __threadfence();
            atomicAdd((unsigned*)gen, 1u);
        } else {
            while (*gen == g) { __builtin_amdgcn_s_sleep(1); }
            __threadfence();
        }
    }
    __syncthreads();
}

// ---------------- prep kernels ----------------
// Fused weights, stored column(K)-major: wc[n*KTOT + k], k<256 -> W_ih[n,k], else W_hh[n,k-256]
__global__ void prep_weights(const float* __restrict__ Wih, const float* __restrict__ Whh,
                             _Float16* __restrict__ wc) {
    int idx = blockIdx.x * 256 + threadIdx.x;
    if (idx >= GATES * KTOT) return;
    int n = idx / KTOT, k = idx % KTOT;
    float v = (k < ISZ) ? Wih[n * ISZ + k] : Whh[n * HID + (k - ISZ)];
    wc[idx] = (_Float16)v;
}

__global__ void prep_x(const float* __restrict__ x, _Float16* __restrict__ xh, int n) {
    for (int i = blockIdx.x * blockDim.x + threadIdx.x; i < n; i += gridDim.x * blockDim.x)
        xh[i] = (_Float16)x[i];
}

__global__ void prep_state(_Float16* __restrict__ hb, unsigned* __restrict__ sy) {
    int i = blockIdx.x * 256 + threadIdx.x;
    if (i < 2 * BATCH * HID) hb[i] = (_Float16)0.0f;
    if (i < 8) sy[i] = 0u;
}

// ---------------- persistent recurrent kernel ----------------
// grid: 32 blocks (htile = channel tile 0..31), block: 128 threads = 4 waves (mtile = batch tile)
// Each wave owns gate tiles {i,f,g,o} x (16 batch rows x 16 channels); c-state stays in VGPRs.
__global__ __launch_bounds__(NTHREADS, 1) void lstm_persist(
        const _Float16* __restrict__ xh,    // [T][64][256] f16
        const _Float16* __restrict__ wc,    // [2048][768] f16, K-major per column
        _Float16* __restrict__ hbuf,        // [2][64][512] f16 (masked h, ping-pong)
        unsigned* __restrict__ sync,        // [0]=cnt, [1]=gen
        const int* __restrict__ keep_mask,  // [512]
        const float* __restrict__ bih, const float* __restrict__ bhh,
        float* __restrict__ out)            // [T][64][512] f32, then h_n [64][512], c_n [64][512]
{
    extern __shared__ _Float16 Bs[];        // [4 gates][16 cols][768 K] f16 = 96 KB

    const int tid   = threadIdx.x;
    const int ln    = tid & 31;
    const int mtile = tid >> 5;             // 0..3
    const int htile = blockIdx.x;           // 0..31
    const int lj    = ln & 15;              // column / A-row within tile
    const int half  = ln >> 4;              // lane half selects K sub-window

    // ---- stage B (this block's 4 gate tiles) into LDS, reused for all 2048 steps ----
    for (int it = tid; it < 4 * 16 * (KTOT / 8); it += NTHREADS) {   // 16-byte chunks
        int gc = it / (KTOT / 8);           // g*16 + c
        int w  = it % (KTOT / 8);
        int g  = gc >> 4, c = gc & 15;
        const uint4* src = (const uint4*)(wc + (size_t)(g * HID + htile * 16 + c) * KTOT + w * 8);
        *((uint4*)(Bs + (size_t)gc * KTOT + w * 8)) = *src;
    }
    __syncthreads();

    // per-lane column constants
    const int j = htile * 16 + lj;
    float bias[4];
#pragma unroll
    for (int g = 0; g < 4; ++g) bias[g] = bih[g * HID + j] + bhh[g * HID + j];
    const float keepf = keep_mask[j] ? 1.0f : 0.0f;

    // persistent cell state (and last h) in registers: rows m = mtile*16 + r + 8*half
    float creg[8], hreg[8];
#pragma unroll
    for (int r = 0; r < 8; ++r) { creg[r] = 0.0f; hreg[r] = 0.0f; }

    const int arow = mtile * 16 + lj;       // A row this lane supplies
    const int aseg = half * 16;             // K sub-offset for A (0 or 16)
    const int bko  = half * 16;             // K sub-offset for B
    volatile unsigned* cnt = sync + 0;
    volatile unsigned* gen = sync + 1;

    // per-lane B column bases in LDS (one per gate)
    const _Float16* b0p = Bs + (size_t)(0 * 16 + lj) * KTOT + bko;
    const _Float16* b1p = Bs + (size_t)(1 * 16 + lj) * KTOT + bko;
    const _Float16* b2p = Bs + (size_t)(2 * 16 + lj) * KTOT + bko;
    const _Float16* b3p = Bs + (size_t)(3 * 16 + lj) * KTOT + bko;

// load chunk kn into buffer set S (no rotation: ping-pong)
#define LOADQ(S, kn)                                                    \
    do {                                                                \
        const _Float16* ap_ = aptr(xrow, hrow, (kn));                   \
        a##S  = ld2(ap_, ap_ + 8);                                      \
        b0##S = ld2(b0p + (kn) * 32, b0p + (kn) * 32 + 8);              \
        b1##S = ld2(b1p + (kn) * 32, b1p + (kn) * 32 + 8);              \
        b2##S = ld2(b2p + (kn) * 32, b2p + (kn) * 32 + 8);              \
        b3##S = ld2(b3p + (kn) * 32, b3p + (kn) * 32 + 8);              \
    } while (0)

#define WMMAQ(S)                                                                                           \
    do {                                                                                                   \
        acc0 = __builtin_amdgcn_wmma_f32_16x16x32_f16(false, a##S, false, b0##S, (short)0, acc0, false, false); \
        acc1 = __builtin_amdgcn_wmma_f32_16x16x32_f16(false, a##S, false, b1##S, (short)0, acc1, false, false); \
        acc2 = __builtin_amdgcn_wmma_f32_16x16x32_f16(false, a##S, false, b2##S, (short)0, acc2, false, false); \
        acc3 = __builtin_amdgcn_wmma_f32_16x16x32_f16(false, a##S, false, b3##S, (short)0, acc3, false, false); \
    } while (0)

    for (int t = 0; t < T_LEN; ++t) {
        const _Float16* hr = hbuf + (size_t)(t & 1) * BATCH * HID;
        _Float16*       hw = hbuf + (size_t)((t + 1) & 1) * BATCH * HID;
        const _Float16* xrow = xh + ((size_t)t * BATCH + arow) * ISZ + aseg;
        const _Float16* hrow = hr + (size_t)arow * HID + aseg;

        if (t + 1 < T_LEN)  // gfx1250 global_prefetch_b8 for next step's activations
            __builtin_prefetch(xh + ((size_t)(t + 1) * BATCH + arow) * ISZ, 0, 1);

        v8f acc0 = {}, acc1 = {}, acc2 = {}, acc3 = {};
        v16h a0, b00, b10, b20, b30;    // ping buffer (even chunks)
        v16h a1, b01, b11, b21, b31;    // pong buffer (odd chunks)

        LOADQ(0, 0);                    // prologue: chunk 0

        // ping-pong pipeline over all 24 chunks: loads one chunk ahead of compute,
        // no inter-buffer copies on the back-edge
#pragma unroll 1
        for (int kc = 0; kc < KCH - 2; kc += 2) {   // kc = 0,2,...,20
            LOADQ(1, kc + 1);
            WMMAQ(0);
            LOADQ(0, kc + 2);
            WMMAQ(1);
            SCHED_PIPE2();
        }
        LOADQ(1, KCH - 1);              // chunk 23
        WMMAQ(0);                       // chunk 22
        WMMAQ(1);                       // chunk 23

        // ---- wave-local LSTM pointwise epilogue ----
#pragma unroll
        for (int r = 0; r < 8; ++r) {
            float gi = acc0[r] + bias[0];
            float gf = acc1[r] + bias[1];
            float gg = acc2[r] + bias[2];
            float go = acc3[r] + bias[3];
            float c_ = sigf(gf) * creg[r] + sigf(gi) * tanh_fast(gg);
            float h_ = sigf(go) * tanh_fast(c_) * keepf;   // mask before feeding next step
            creg[r] = c_;
            hreg[r] = h_;
            int m = mtile * 16 + r + half * 8;             // C/D tile row mapping
            // out is write-once: non-temporal keeps L2 for weights/x/h working set
            __builtin_nontemporal_store(h_, &out[((size_t)t * BATCH + m) * HID + j]);
            hw[m * HID + j] = (_Float16)h_;
        }

        grid_sync(cnt, gen);   // h(t) visible everywhere before step t+1's GEMM
    }

    // final states: h_n then c_n appended after output
#pragma unroll
    for (int r = 0; r < 8; ++r) {
        int m = mtile * 16 + r + half * 8;
        out[(size_t)T_LEN * BATCH * HID + (size_t)m * HID + j] = hreg[r];
        out[(size_t)T_LEN * BATCH * HID + (size_t)BATCH * HID + (size_t)m * HID + j] = creg[r];
    }
#undef LOADQ
#undef WMMAQ
}

// ---------------- launch ----------------
extern "C" void kernel_launch(void* const* d_in, const int* in_sizes, int n_in,
                              void* d_out, int out_size, void* d_ws, size_t ws_size,
                              hipStream_t stream) {
    (void)in_sizes; (void)n_in; (void)out_size; (void)ws_size;
    const float* x   = (const float*)d_in[0];
    const int*  keep = (const int*)d_in[1];
    const float* Wih = (const float*)d_in[2];
    const float* Whh = (const float*)d_in[3];
    const float* bih = (const float*)d_in[4];
    const float* bhh = (const float*)d_in[5];
    float* out = (float*)d_out;

    char* ws = (char*)d_ws;
    _Float16* wc = (_Float16*)(ws + WC_OFF);
    _Float16* xh = (_Float16*)(ws + XH_OFF);
    _Float16* hb = (_Float16*)(ws + HB_OFF);
    unsigned* sy = (unsigned*)(ws + SY_OFF);

    prep_weights<<<(GATES * KTOT + 255) / 256, 256, 0, stream>>>(Wih, Whh, wc);
    prep_x<<<2048, 256, 0, stream>>>(x, xh, T_LEN * BATCH * ISZ);
    prep_state<<<(2 * BATCH * HID + 255) / 256, 256, 0, stream>>>(hb, sy);

    const size_t lds_bytes = (size_t)4 * 16 * KTOT * sizeof(_Float16);   // 96 KB
    (void)hipFuncSetAttribute((const void*)lstm_persist,
                              hipFuncAttributeMaxDynamicSharedMemorySize, (int)lds_bytes);
    lstm_persist<<<NBLOCKS, NTHREADS, lds_bytes, stream>>>(xh, wc, hb, sy, keep, bih, bhh, out);
}
